// HeterogeneousNCA_31963146617163
// MI455X (gfx1250) — matrix operand order repaired and checked by reading, not compiled, not run on hardware
//
#include <hip/hip_runtime.h>
#include <hip/hip_bf16.h>

typedef __attribute__((ext_vector_type(16))) _Float16 v16h;
typedef __attribute__((ext_vector_type(8)))  _Float16 v8h;
typedef __attribute__((ext_vector_type(8)))  float    v8f;
typedef __attribute__((ext_vector_type(4)))  float    v4f;

#define BATCH 4
#define CC 64           // channel stride for all GEMM A tensors (padded)
#define HW 4096
#define NCH 11
#define HEADS 4
#define HD 16

// ---------------------------------------------------------------------------
// A-fragment (16x32 f16, M=pixels, K=channels), doc layout:
// lane l: m = l&15, hh = l>>4 ; vgpr v: k = (v<4?0:16) + hh*8 + (v&3)*2
// => per lane the fragment is two contiguous 16B chunks:
//    halfs [kc*32 + hh*8 .. +7] and [kc*32 + 16 + hh*8 .. +7].
// Boundary pixels: clamp address, zero via register select (no exec dance).
// ---------------------------------------------------------------------------
template <int CONV3>
__device__ inline v16h load_a_frag(const _Float16* __restrict__ Ab, int posbase,
                                   int lane, int dy, int dx, int kc) {
  int m = lane & 15, hh = lane >> 4;
  int p = posbase + m;
  int yy = (p >> 6), xx = (p & 63);
  if (CONV3) { yy += dy - 1; xx += dx - 1; }
  bool inb = !CONV3 || (((unsigned)yy < 64u) && ((unsigned)xx < 64u));
  int pix = inb ? (yy * 64 + xx) : 0;
  const _Float16* row = Ab + (size_t)pix * CC + kc * 32 + hh * 8;
  v8h lo = *(const v8h*)(row);
  v8h hi = *(const v8h*)(row + 16);
  v8h z = {};
  lo = inb ? lo : z;
  hi = inb ? hi : z;
  v16h a;
#pragma unroll
  for (int e = 0; e < 8; ++e) { a[e] = lo[e]; a[8 + e] = hi[e]; }
  return a;
}

// ---------------------------------------------------------------------------
// Generic implicit-GEMM conv (3x3 SAME or 1x1) via WMMA f16->f32.
// Weights pre-packed in B-fragment order: (((tap*2+kc)*NT+nc)*32+lane)*16+e,
// with k = kc*32 + (e>>3)*16 + (lane>>4)*8 + (e&7), n = nc*16 + (lane&15).
// epi: 0=none 1=relu 2=sigmoid. nchw: write NCHW fp32 (decoder out).
// ---------------------------------------------------------------------------
template <int TAPS, int NT>
__global__ void gemm_conv(const _Float16* __restrict__ A,
                          const _Float16* __restrict__ Wpk,
                          const float* __restrict__ bias,
                          float* __restrict__ outF, _Float16* __restrict__ outH,
                          int Cout, int ostride, int epi, int nchw) {
  int g = blockIdx.x;                      // BATCH*256 m-tiles
  int b = g >> 8;
  int posbase = (g & 255) << 4;
  int wave = threadIdx.x >> 5, lane = threadIdx.x & 31;
  int hh = lane >> 4;
  const _Float16* Ab = A + (size_t)b * HW * CC;
  const v16h* wbase = (const v16h*)Wpk;

  v16h a01[2];
  if (TAPS == 1) {
    a01[0] = load_a_frag<0>(Ab, posbase, lane, 0, 0, 0);
    a01[1] = load_a_frag<0>(Ab, posbase, lane, 0, 0, 1);
  }

  for (int nc = wave; nc < NT; nc += 4) {
    v8f acc = {};
#pragma unroll
    for (int t = 0; t < TAPS; ++t) {
#pragma unroll
      for (int kc = 0; kc < 2; ++kc) {
        v16h a = (TAPS == 1) ? a01[kc]
                             : load_a_frag<1>(Ab, posbase, lane, t / 3, t % 3, kc);
        const v16h* wp = wbase + (((size_t)(t * 2 + kc) * NT + nc) * 32 + lane);
        __builtin_prefetch(wp + 32, 0, 1);
        v16h bf = *wp;
        acc = __builtin_amdgcn_wmma_f32_16x16x32_f16(false, a, false, bf,
                                                     (short)0, acc, false, false);
      }
    }
    int n = nc * 16 + (lane & 15);
    if (n < Cout) {
      float bi = bias[n];
#pragma unroll
      for (int v = 0; v < 8; ++v) {
        int m = v + 8 * hh;
        float val = acc[v] + bi;
        if (epi == 1) val = fmaxf(val, 0.0f);
        else if (epi == 2) val = 1.0f / (1.0f + __expf(-val));
        size_t oidx;
        if (nchw) oidx = ((size_t)(b * Cout + n)) * HW + posbase + m;
        else      oidx = ((size_t)(b * HW + posbase + m)) * ostride + n;
        if (outF) outF[oidx] = val;
        if (outH) outH[oidx] = (_Float16)val;
      }
    }
  }
}

// ---------------------------------------------------------------------------
// Pack fp32 weights (Cout, Cin, kh, kw) into f16 B-fragment tiles.
// ---------------------------------------------------------------------------
__global__ void pack_w(const float* __restrict__ w, _Float16* __restrict__ dst,
                       int Cout, int Cin, int taps, int NT) {
  int idx = blockIdx.x * blockDim.x + threadIdx.x;
  int total = taps * 2 * NT * 512;
  if (idx >= total) return;
  int e = idx & 15;
  int lane = (idx >> 4) & 31;
  int rest = idx >> 9;
  int nc = rest % NT;
  int kcc = rest / NT;
  int kc = kcc & 1;
  int t  = kcc >> 1;
  int n = nc * 16 + (lane & 15);
  int k = kc * 32 + (e >> 3) * 16 + (lane >> 4) * 8 + (e & 7);
  _Float16 val = (_Float16)0;
  if (n < Cout && k < Cin) val = (_Float16)w[((size_t)(n * Cin + k)) * taps + t];
  dst[idx] = val;
}

// ---------------------------------------------------------------------------
// Build encoder input: channel-last f16 (B, 4096, 64): [x(11) | task_emb(32) | 0]
// ---------------------------------------------------------------------------
__global__ void build_enc_in(const float* __restrict__ x,
                             const float* __restrict__ temb,
                             _Float16* __restrict__ dst) {
  int idx = blockIdx.x * blockDim.x + threadIdx.x;
  if (idx >= BATCH * HW * CC) return;
  int c = idx & 63;
  int p = (idx >> 6) & (HW - 1);
  int b = idx >> 18;
  float v = 0.0f;
  if (c < NCH)      v = x[((size_t)(b * NCH + c)) * HW + p];
  else if (c < 43)  v = temb[b * 32 + (c - NCH)];
  dst[idx] = (_Float16)v;
}

// ---------------------------------------------------------------------------
// Instance norm: one block per (b,c), population variance, eps=1e-5.
// ---------------------------------------------------------------------------
__global__ void inorm(const float* __restrict__ state, _Float16* __restrict__ nrm) {
  int b = blockIdx.x >> 6;
  int c = blockIdx.x & 63;
  __shared__ float s1[256], s2[256];
  const float* base = state + (size_t)b * HW * CC + c;
  float sum = 0.f, sq = 0.f;
  for (int p = threadIdx.x; p < HW; p += 256) {
    float v = base[(size_t)p * CC];
    sum += v; sq += v * v;
  }
  s1[threadIdx.x] = sum; s2[threadIdx.x] = sq;
  __syncthreads();
  for (int s = 128; s > 0; s >>= 1) {
    if (threadIdx.x < s) { s1[threadIdx.x] += s1[threadIdx.x + s];
                           s2[threadIdx.x] += s2[threadIdx.x + s]; }
    __syncthreads();
  }
  float mu = s1[0] * (1.0f / HW);
  float var = s2[0] * (1.0f / HW) - mu * mu;
  float inv = rsqrtf(var + 1e-5f);
  _Float16* nb = nrm + (size_t)b * HW * CC + c;
  for (int p = threadIdx.x; p < HW; p += 256)
    nb[(size_t)p * CC] = (_Float16)((base[(size_t)p * CC] - mu) * inv);
}

// ---------------------------------------------------------------------------
// Flash attention, one wave per 16-query tile, streaming 16-key blocks.
// qkv channel-last (B, 4096, 192): [q(64)|k(64)|v(64)], head h uses dims h*16..+15.
// - V tiles: double-buffered CDNA5 async global->LDS copies (software pipeline
//   on ASYNCcnt; s_wait_asynccnt 1 retires only the tile being consumed).
// - Softmax: row-ownership through a per-wave LDS S-tile (batched ds_b128
//   traffic) instead of ds_bpermute butterflies.
// ---------------------------------------------------------------------------
__global__ void flash_attn(const _Float16* __restrict__ qkv,
                           _Float16* __restrict__ out) {
  int bh = blockIdx.x >> 6;                 // 0..15
  int b = bh >> 2, h = bh & 3;
  int wid = threadIdx.x >> 5;
  int itile = ((blockIdx.x & 63) << 2) + wid;
  int ibase = itile << 4;
  int lane = threadIdx.x & 31;
  int m = lane & 15, hh = lane >> 4;

  __shared__ __align__(64) float    st_[4][256];     // S tile (f32, i x j)
  __shared__ __align__(32) _Float16 pt_[4][256];     // P tile (f16, i x j)
  __shared__ __align__(32) _Float16 vt_[4][2][256];  // V tiles (double buffer)
  __shared__ __align__(32) float    sc_[4][16];      // per-row rescale
  __shared__ __align__(32) float    lr_[4][16];      // per-row l (final)
  float*    st  = st_[wid];
  _Float16* myp = pt_[wid];
  _Float16* mv0 = vt_[wid][0];
  _Float16* mv1 = vt_[wid][1];
  float*    scr = sc_[wid];
  float*    lrr = lr_[wid];

  const _Float16* base = qkv + (size_t)b * HW * 192;
  const _Float16* Q = base + h * HD;
  const _Float16* K = base + 64 + h * HD;
  const _Float16* V = base + 128 + h * HD;

  // Q fragment (A layout, K padded 16->32), pre-scaled by HD^-0.5 = 0.25.
  v16h qf;
  {
    const _Float16* qrow = Q + (size_t)(ibase + m) * 192;
    v8h q0 = *(const v8h*)(qrow + hh * 8);
#pragma unroll
    for (int e = 0; e < 8; ++e) {
      qf[e]     = (_Float16)((float)q0[e] * 0.25f);
      qf[8 + e] = (_Float16)0;
    }
  }

  // async V-tile staging: lane covers one 16B chunk (row lane>>1, col (lane&1)*8)
  int vrow = lane >> 1, vcol = (lane & 1) * 8;
  unsigned lds0 = (unsigned)(unsigned long long)(mv0 + vrow * 16 + vcol);
  unsigned lds1 = (unsigned)(unsigned long long)(mv1 + vrow * 16 + vcol);
  const _Float16* vsrc0 = V + (size_t)vrow * 192 + vcol;

  // prologue: kick tile 0 into buffer 0
  {
    unsigned long long gsrc = (unsigned long long)vsrc0;
    asm volatile("global_load_async_to_lds_b128 %0, %1, off"
                 :: "v"(lds0), "v"(gsrc) : "memory");
  }

  v8f o = {};
  float mrow = -1e30f, lrow = 0.0f;        // row (lane&15) running stats

  for (int jb = 0; jb < HW; jb += 16) {
    int cur = (jb >> 4) & 1;
    bool more = (jb + 16) < HW;
    if (more) {   // pipeline: start next tile into the other buffer
      unsigned long long gs = (unsigned long long)(vsrc0 + (size_t)(jb + 16) * 192);
      asm volatile("global_load_async_to_lds_b128 %0, %1, off"
                   :: "v"(cur ? lds0 : lds1), "v"(gs) : "memory");
    }

    // K block fragment (B layout): half-group hh holds k=hh*8..+7 in elems 0..7; elems 8..15 zero-pad
    v8h kr = *(const v8h*)(K + (size_t)(jb + m) * 192 + hh * 8);
    v16h kf;
#pragma unroll
    for (int e = 0; e < 8; ++e) { kf[e] = kr[e]; kf[8 + e] = (_Float16)0; }
    v8f zc = {};
    v8f s = __builtin_amdgcn_wmma_f32_16x16x32_f16(false, qf, false, kf,
                                                   (short)0, zc, false, false);
    // scatter S fragment to LDS tile (i = v + 8*hh, j = lane&15)
#pragma unroll
    for (int v = 0; v < 8; ++v) st[(v + 8 * hh) * 16 + m] = s[v];

    // row ownership: lane handles row r = lane&15 (both half-groups redundant)
    float rowv[16];
    {
      v4f r0 = *(const v4f*)(st + m * 16 + 0);
      v4f r1 = *(const v4f*)(st + m * 16 + 4);
      v4f r2 = *(const v4f*)(st + m * 16 + 8);
      v4f r3 = *(const v4f*)(st + m * 16 + 12);
#pragma unroll
      for (int e = 0; e < 4; ++e) {
        rowv[e] = r0[e]; rowv[4 + e] = r1[e];
        rowv[8 + e] = r2[e]; rowv[12 + e] = r3[e];
      }
    }
    float mx = rowv[0];
#pragma unroll
    for (int e = 1; e < 16; ++e) mx = fmaxf(mx, rowv[e]);
    float mnew = fmaxf(mrow, mx);
    float scx = __expf(mrow - mnew);
    float rsum = 0.0f;
    float pv16[16];
#pragma unroll
    for (int e = 0; e < 16; ++e) { pv16[e] = __expf(rowv[e] - mnew); rsum += pv16[e]; }
    mrow = mnew;
    lrow = lrow * scx + rsum;
    // write this half-group's half of the P row (16B contiguous)
    v8h phl, phh;
#pragma unroll
    for (int e = 0; e < 8; ++e) { phl[e] = (_Float16)pv16[e]; phh[e] = (_Float16)pv16[8 + e]; }
    v8h ph = hh ? phh : phl;
    *(v8h*)(myp + m * 16 + hh * 8) = ph;
    scr[m] = scx;                          // duplicate same-value writes OK

    // rescale O accumulator: this lane's rows are 8*hh + v (contiguous scales)
    {
      v4f s0 = *(const v4f*)(scr + 8 * hh);
      v4f s1 = *(const v4f*)(scr + 8 * hh + 4);
#pragma unroll
      for (int v = 0; v < 4; ++v) { o[v] *= s0[v]; o[4 + v] *= s1[v]; }
    }
    // P A-fragment: one contiguous ds_load_b128 per lane, upper K half zero
    v16h pf;
    {
      v8h prow = *(const v8h*)(myp + m * 16 + hh * 8);
#pragma unroll
      for (int e = 0; e < 8; ++e) { pf[e] = prow[e]; pf[8 + e] = (_Float16)0; }
    }
    // retire the tile we are about to consume (keep the next one in flight)
    if (more) asm volatile("s_wait_asynccnt 0x1" ::: "memory");
    else      asm volatile("s_wait_asynccnt 0x0" ::: "memory");
    const _Float16* vb = cur ? mv1 : mv0;
    v16h vf;
#pragma unroll
    for (int e = 0; e < 16; ++e) {
      vf[e] = (e < 8) ? vb[(hh * 8 + e) * 16 + m] : (_Float16)0;
    }
    o = __builtin_amdgcn_wmma_f32_16x16x32_f16(false, pf, false, vf,
                                               (short)0, o, false, false);
  }

  // final normalization: per-row l through LDS (contiguous per half-group)
  lrr[m] = lrow;
  v4f l0 = *(const v4f*)(lrr + 8 * hh);
  v4f l1 = *(const v4f*)(lrr + 8 * hh + 4);
  _Float16* ob = out + ((size_t)(b * HW + ibase)) * CC + h * HD;
#pragma unroll
  for (int v = 0; v < 8; ++v) {
    int i = v + 8 * hh;
    float ln = (v < 4) ? l0[v & 3] : l1[v & 3];
    ob[(size_t)i * CC + m] = (_Float16)(o[v] / ln);
  }
}

// ---------------------------------------------------------------------------
// Fused per-pixel update: mix-MLP -> alpha, gate accumulation, state update.
// Mix weights staged in LDS once per block.
// ---------------------------------------------------------------------------
__global__ void update_state(const float* __restrict__ localf,
                             const float* __restrict__ globalf,
                             const float* __restrict__ betaf,
                             const float* __restrict__ w1, const float* __restrict__ b1,
                             const float* __restrict__ w2, const float* __restrict__ b2,
                             float* __restrict__ state, _Float16* __restrict__ stateh,
                             float* __restrict__ gate_acc) {
  __shared__ float w1s[2048];
  __shared__ float b1s[16], w2s[16];
  for (int i = threadIdx.x; i < 2048; i += 256) w1s[i] = w1[i];
  if (threadIdx.x < 16) { b1s[threadIdx.x] = b1[threadIdx.x];
                          w2s[threadIdx.x] = w2[threadIdx.x]; }
  __syncthreads();

  int idx = blockIdx.x * blockDim.x + threadIdx.x;
  if (idx >= BATCH * HW) return;
  const float* lv = localf  + (size_t)idx * CC;
  const float* gv = globalf + (size_t)idx * CC;
  float z = b2[0];
  for (int i = 0; i < 16; ++i) {
    float hsum = b1s[i];
    const float* wrow = w1s + i * 128;
#pragma unroll 8
    for (int c = 0; c < 64; ++c) hsum += wrow[c] * lv[c];
#pragma unroll 8
    for (int c = 0; c < 64; ++c) hsum += wrow[64 + c] * gv[c];
    z += w2s[i] * fmaxf(hsum, 0.0f);
  }
  float alpha = 1.0f / (1.0f + __expf(-z));
  atomicAdd(gate_acc, alpha);
  float* sv = state + (size_t)idx * CC;
  _Float16* sh = stateh + (size_t)idx * CC;
  const float* bv = betaf + (size_t)idx * CC;
  for (int c = 0; c < 64; ++c) {
    float delta = alpha * lv[c] + (1.0f - alpha) * gv[c];
    float beta  = bv[c];
    float ns = beta * sv[c] + (1.0f - beta) * delta;
    sv[c] = ns;
    sh[c] = (_Float16)ns;
  }
}

__global__ void zero_gates(float* g) { if (threadIdx.x < 5) g[threadIdx.x] = 0.0f; }

__global__ void finalize_gates(const float* __restrict__ g, float* __restrict__ out) {
  if (threadIdx.x < 5) out[threadIdx.x] = g[threadIdx.x] * (1.0f / (BATCH * HW));
}

// ---------------------------------------------------------------------------
extern "C" void kernel_launch(void* const* d_in, const int* in_sizes, int n_in,
                              void* d_out, int out_size, void* d_ws, size_t ws_size,
                              hipStream_t stream) {
  const float* x       = (const float*)d_in[0];
  const float* temb    = (const float*)d_in[1];
  const float* enc_w1  = (const float*)d_in[2];
  const float* enc_b1  = (const float*)d_in[3];
  const float* enc_w2  = (const float*)d_in[4];
  const float* enc_b2  = (const float*)d_in[5];
  const float* loc_w1  = (const float*)d_in[6];
  const float* loc_b1  = (const float*)d_in[7];
  const float* loc_w2  = (const float*)d_in[8];
  const float* loc_b2  = (const float*)d_in[9];
  const float* qkv_w   = (const float*)d_in[10];
  const float* qkv_b   = (const float*)d_in[11];
  const float* proj_w  = (const float*)d_in[12];
  const float* proj_b  = (const float*)d_in[13];
  const float* mix_w1  = (const float*)d_in[14];
  const float* mix_b1  = (const float*)d_in[15];
  const float* mix_w2  = (const float*)d_in[16];
  const float* mix_b2  = (const float*)d_in[17];
  const float* tau_w   = (const float*)d_in[18];
  const float* tau_b   = (const float*)d_in[19];
  const float* dec_w1  = (const float*)d_in[20];
  const float* dec_b1  = (const float*)d_in[21];
  const float* dec_w2  = (const float*)d_in[22];
  const float* dec_b2  = (const float*)d_in[23];
  float* outp = (float*)d_out;

  char* wsb = (char*)d_ws;
  size_t off = 0;
  auto alloc = [&](size_t bytes) -> char* {
    char* p = wsb + off;
    off += (bytes + 255) & ~(size_t)255;
    return p;
  };
  const size_t ACT_H = (size_t)BATCH * HW * CC * sizeof(_Float16);
  const size_t ACT_F = (size_t)BATCH * HW * CC * sizeof(float);

  _Float16* ENC_IN  = (_Float16*)alloc(ACT_H);
  float*    STATE_F = (float*)   alloc(ACT_F);
  _Float16* STATE_H = (_Float16*)alloc(ACT_H);
  _Float16* T1_H    = (_Float16*)alloc(ACT_H);
  float*    LOCAL_F = (float*)   alloc(ACT_F);
  _Float16* NORM_H  = (_Float16*)alloc(ACT_H);
  _Float16* QKV_H   = (_Float16*)alloc((size_t)BATCH * HW * 192 * sizeof(_Float16));
  _Float16* ATTN_H  = (_Float16*)alloc(ACT_H);
  float*    GLOB_F  = (float*)   alloc(ACT_F);
  float*    BETA_F  = (float*)   alloc(ACT_F);
  _Float16* D1_H    = (_Float16*)alloc(ACT_H);
  float*    GACC    = (float*)   alloc(64);
  _Float16* PK_ENC1 = (_Float16*)alloc(9 * 2 * 4 * 512 * 2);
  _Float16* PK_ENC2 = (_Float16*)alloc(9 * 2 * 4 * 512 * 2);
  _Float16* PK_LOC1 = (_Float16*)alloc(9 * 2 * 4 * 512 * 2);
  _Float16* PK_LOC2 = (_Float16*)alloc(1 * 2 * 4 * 512 * 2);
  _Float16* PK_QKV  = (_Float16*)alloc(1 * 2 * 12 * 512 * 2);
  _Float16* PK_PROJ = (_Float16*)alloc(1 * 2 * 4 * 512 * 2);
  _Float16* PK_TAU  = (_Float16*)alloc(1 * 2 * 4 * 512 * 2);
  _Float16* PK_DEC1 = (_Float16*)alloc(9 * 2 * 4 * 512 * 2);
  _Float16* PK_DEC2 = (_Float16*)alloc(1 * 2 * 1 * 512 * 2);
  (void)ws_size; (void)n_in; (void)in_sizes; (void)out_size;

  zero_gates<<<1, 32, 0, stream>>>(GACC);
  build_enc_in<<<(BATCH * HW * CC + 255) / 256, 256, 0, stream>>>(x, temb, ENC_IN);

  auto pack = [&](const float* w, _Float16* dst, int Cout, int Cin, int taps, int NT) {
    int total = taps * 2 * NT * 512;
    pack_w<<<(total + 255) / 256, 256, 0, stream>>>(w, dst, Cout, Cin, taps, NT);
  };
  pack(enc_w1, PK_ENC1, 64, 43, 9, 4);
  pack(enc_w2, PK_ENC2, 64, 64, 9, 4);
  pack(loc_w1, PK_LOC1, 64, 64, 9, 4);
  pack(loc_w2, PK_LOC2, 64, 64, 1, 4);
  pack(qkv_w,  PK_QKV, 192, 64, 1, 12);
  pack(proj_w, PK_PROJ, 64, 64, 1, 4);
  pack(tau_w,  PK_TAU,  64, 64, 1, 4);
  pack(dec_w1, PK_DEC1, 64, 64, 9, 4);
  pack(dec_w2, PK_DEC2, NCH, 64, 1, 1);

  const int GEMM_GRID = BATCH * 256;

  // encoder
  gemm_conv<9, 4><<<GEMM_GRID, 128, 0, stream>>>(ENC_IN, PK_ENC1, enc_b1,
      nullptr, T1_H, 64, 64, 1, 0);
  gemm_conv<9, 4><<<GEMM_GRID, 128, 0, stream>>>(T1_H, PK_ENC2, enc_b2,
      STATE_F, STATE_H, 64, 64, 1, 0);

  for (int step = 0; step < 5; ++step) {
    gemm_conv<9, 4><<<GEMM_GRID, 128, 0, stream>>>(STATE_H, PK_LOC1, loc_b1,
        nullptr, T1_H, 64, 64, 1, 0);
    gemm_conv<1, 4><<<GEMM_GRID, 128, 0, stream>>>(T1_H, PK_LOC2, loc_b2,
        LOCAL_F, nullptr, 64, 64, 0, 0);
    inorm<<<256, 256, 0, stream>>>(STATE_F, NORM_H);
    gemm_conv<1, 12><<<GEMM_GRID, 128, 0, stream>>>(NORM_H, PK_QKV, qkv_b,
        nullptr, QKV_H, 192, 192, 0, 0);
    flash_attn<<<1024, 128, 0, stream>>>(QKV_H, ATTN_H);
    gemm_conv<1, 4><<<GEMM_GRID, 128, 0, stream>>>(ATTN_H, PK_PROJ, proj_b,
        GLOB_F, nullptr, 64, 64, 0, 0);
    gemm_conv<1, 4><<<GEMM_GRID, 128, 0, stream>>>(STATE_H, PK_TAU, tau_b,
        BETA_F, nullptr, 64, 64, 2, 0);
    update_state<<<(BATCH * HW + 255) / 256, 256, 0, stream>>>(
        LOCAL_F, GLOB_F, BETA_F, mix_w1, mix_b1, mix_w2, mix_b2,
        STATE_F, STATE_H, GACC + step);
  }

  // decoder
  gemm_conv<9, 4><<<GEMM_GRID, 128, 0, stream>>>(STATE_H, PK_DEC1, dec_b1,
      nullptr, D1_H, 64, 64, 1, 0);
  gemm_conv<1, 1><<<GEMM_GRID, 128, 0, stream>>>(D1_H, PK_DEC2, dec_b2,
      outp, nullptr, NCH, NCH, 0, 1);

  finalize_gates<<<1, 32, 0, stream>>>(GACC, outp + (size_t)BATCH * NCH * HW);
}
